// MultiHeadAttention_26456998543367
// MI455X (gfx1250) — compile-verified, matrix operand order checked by reading, not run on hardware
//
#include <hip/hip_runtime.h>
#include <hip/hip_bf16.h>

// ---------------------------------------------------------------------------
// MI455X (gfx1250) multi-head attention with relative position bias.
// All matmuls via v_wmma_f32_16x16x32_bf16 (f32 accumulation).
// ---------------------------------------------------------------------------

typedef __bf16 bf16;
typedef __attribute__((ext_vector_type(16))) __bf16 v16bf;
typedef __attribute__((ext_vector_type(8)))  __bf16 v8bf;
typedef __attribute__((ext_vector_type(8)))  float  v8f;

#define DM    1024
#define NH    16
#define HD    64
#define LSEQ  2048
#define BATCH 2
#define GM    (BATCH*LSEQ)   /* 4096 rows of x */
#define RELROWS (2*2048-1)   /* 4095 */

// Fragment loader: 8+8 contiguous bf16 -> v16bf (per-lane WMMA fragment slice).
static __device__ __forceinline__ v16bf ld_frag(const bf16* p0, const bf16* p1) {
  v8bf a = *(const v8bf*)p0;
  v8bf b = *(const v8bf*)p1;
  v16bf r;
#pragma unroll
  for (int i = 0; i < 8; ++i) { r[i] = a[i]; r[i + 8] = b[i]; }
  return r;
}

static __device__ __forceinline__ v8f wmma_bf16(v16bf a, v16bf b, v8f c) {
  // 8 args: (neg_a, A, neg_b, B, c_mod, C, reuse_a, reuse_b)
  return __builtin_amdgcn_wmma_f32_16x16x32_bf16(false, a, false, b, (short)0, c,
                                                 false, false);
}

// ---------------------------------------------------------------------------
// Stage 1: f32 -> bf16 conversion
// ---------------------------------------------------------------------------
__global__ void cvt_f32_bf16(const float* __restrict__ src, bf16* __restrict__ dst,
                             int n) {
  int i = blockIdx.x * blockDim.x + threadIdx.x;
  if (i < n) dst[i] = (bf16)src[i];
}

// ---------------------------------------------------------------------------
// Stage 2/4: GEMM  y[m,n] = sum_k A[m,k] * W[n,k] + bias[n]
//   A: [4096,1024] bf16 row-major, W: [1024,1024] bf16 row-major (torch Linear).
//   MODE 0: write bf16 head-split [B,H,L,64]
//   MODE 1: write f32 row-major [4096,1024] (final output)
// Block: 256 thr (8 waves). Tile 128x128. Wave: 64x32 (4x2 WMMA tiles). K-step 32.
// ---------------------------------------------------------------------------
template <int MODE>
__global__ __launch_bounds__(256)
void gemm_wmma(const bf16* __restrict__ A, const bf16* __restrict__ W,
               const float* __restrict__ bias, void* __restrict__ dst) {
  __shared__ bf16 As[128 * 32];
  __shared__ bf16 Bs[128 * 32];

  const int tid   = threadIdx.x;
  const int wave  = tid >> 5;
  const int lane  = tid & 31;
  const int half  = lane >> 4;       // WMMA fragment half (lanes 0-15 / 16-31)
  const int lanen = lane & 15;
  const int kb0   = half ? 8 : 0;    // per-lane K sub-offsets within a 32-chunk
  const int kb1   = kb0 + 16;

  const int m0 = blockIdx.y * 128;
  const int n0 = blockIdx.x * 128;
  const int waveM = wave >> 2;       // 0..1  -> 64-row strip
  const int waveN = wave & 3;        // 0..3  -> 32-col strip

  v8f acc[4][2];
#pragma unroll
  for (int i = 0; i < 4; ++i)
#pragma unroll
    for (int j = 0; j < 2; ++j)
#pragma unroll
      for (int e = 0; e < 8; ++e) acc[i][j][e] = 0.0f;

  // cooperative tile load mapping: 2 threads per 32-wide row
  const int lr = tid >> 1;
  const int lc = (tid & 1) * 16;

  for (int k0 = 0; k0 < DM; k0 += 32) {
    __syncthreads();
    {
      const bf16* sa = A + (size_t)(m0 + lr) * DM + k0 + lc;
      *(v8bf*)(As + lr * 32 + lc)     = *(const v8bf*)(sa);
      *(v8bf*)(As + lr * 32 + lc + 8) = *(const v8bf*)(sa + 8);
      const bf16* sb = W + (size_t)(n0 + lr) * DM + k0 + lc;
      *(v8bf*)(Bs + lr * 32 + lc)     = *(const v8bf*)(sb);
      *(v8bf*)(Bs + lr * 32 + lc + 8) = *(const v8bf*)(sb + 8);
      if (k0 + 32 < DM) {  // pull next K-tile toward L2 (global_prefetch_b8)
        __builtin_prefetch(A + (size_t)(m0 + lr) * DM + k0 + 32 + lc, 0, 1);
        __builtin_prefetch(W + (size_t)(n0 + lr) * DM + k0 + 32 + lc, 0, 1);
      }
    }
    __syncthreads();

    v16bf af[4], bfr[2];
#pragma unroll
    for (int i = 0; i < 4; ++i) {
      const bf16* p = As + (waveM * 64 + i * 16 + lanen) * 32;
      af[i] = ld_frag(p + kb0, p + kb1);
    }
#pragma unroll
    for (int j = 0; j < 2; ++j) {
      const bf16* p = Bs + (waveN * 32 + j * 16 + lanen) * 32;
      bfr[j] = ld_frag(p + kb0, p + kb1);
    }
#pragma unroll
    for (int i = 0; i < 4; ++i)
#pragma unroll
      for (int j = 0; j < 2; ++j) acc[i][j] = wmma_bf16(af[i], bfr[j], acc[i][j]);
  }

  // epilogue: C-frag element (row = v + 8*half, col = lanen)
#pragma unroll
  for (int j = 0; j < 2; ++j) {
    const int n_g = n0 + waveN * 32 + j * 16 + lanen;
    const float bj = bias[n_g];
#pragma unroll
    for (int i = 0; i < 4; ++i) {
      const int m_base = m0 + waveM * 64 + i * 16;
#pragma unroll
      for (int v = 0; v < 8; ++v) {
        const int   m_g = m_base + v + 8 * half;
        const float val = acc[i][j][v] + bj;
        if (MODE == 0) {
          const int b = m_g >> 11, l = m_g & 2047;
          const int h = n_g >> 6,  d = n_g & 63;
          ((bf16*)dst)[(((size_t)(b * NH + h) * LSEQ + l) * HD) + d] = (bf16)val;
        } else {
          ((float*)dst)[(size_t)m_g * DM + n_g] = val;
        }
      }
    }
  }
}

// ---------------------------------------------------------------------------
// Stage 3: flash attention with relative bias.
// grid = (L/128, H, B), block = 256 (8 waves). Wave w owns Q rows
// [l0, l0+16). Key tiles of 32 streamed through LDS. Per key tile:
//   6 WMMA: P = Q @ rel_band^T (16x48, skewed gather gives rel bias)
//   4 WMMA: S = Q @ K^T
//   4 WMMA: O += softmax(S) @ V
// ---------------------------------------------------------------------------
__global__ __launch_bounds__(256)
void attn_wmma(const bf16* __restrict__ q, const bf16* __restrict__ k,
               const bf16* __restrict__ v, const bf16* __restrict__ rel,
               bf16* __restrict__ o) {
  __shared__ bf16  Ks[32 * 64];        // K tile, row-major [key][d]
  __shared__ bf16  Vt[64 * 32];        // V tile transposed  [d][key]
  __shared__ float Ps[8][16 * 48];     // per-wave scratch (P band / att staging)

  const int tid   = threadIdx.x;
  const int wave  = tid >> 5;
  const int lane  = tid & 31;
  const int half  = lane >> 4;
  const int lanen = lane & 15;
  const int kb0   = half ? 8 : 0;
  const int kb1   = kb0 + 16;

  const int b  = blockIdx.z;
  const int h  = blockIdx.y;
  const int l0 = blockIdx.x * 128 + wave * 16;
  const size_t bh = (size_t)(b * NH + h) * LSEQ * HD;

  // Q A-fragments (16 rows x 64 d = two K-chunks of 32), held in registers.
  v16bf aq[2];
#pragma unroll
  for (int c = 0; c < 2; ++c) {
    const bf16* qp = q + bh + (size_t)(l0 + lanen) * HD + c * 32;
    aq[c] = ld_frag(qp + kb0, qp + kb1);
  }

  float mrow[8], lsum[8];
  v8f   oacc[4];
#pragma unroll
  for (int e = 0; e < 8; ++e) { mrow[e] = -1e30f; lsum[e] = 0.0f; }
#pragma unroll
  for (int nc = 0; nc < 4; ++nc)
#pragma unroll
    for (int e = 0; e < 8; ++e) oacc[nc][e] = 0.0f;

  float* myPs = Ps[wave];
  const int loadrow = tid >> 3;        // 0..31 key row
  const int loadcol = (tid & 7) * 8;   // 8 bf16 per thread

  for (int r0 = 0; r0 < LSEQ; r0 += 32) {
    __syncthreads();
    {
      const bf16* kp = k + bh + (size_t)(r0 + loadrow) * HD + loadcol;
      *(v8bf*)(Ks + loadrow * 64 + loadcol) = *(const v8bf*)kp;
      const bf16* vp = v + bh + (size_t)(r0 + loadrow) * HD + loadcol;
      v8bf vv = *(const v8bf*)vp;
#pragma unroll
      for (int i = 0; i < 8; ++i) Vt[(loadcol + i) * 32 + loadrow] = vv[i];
    }
    __syncthreads();

    // --- relative-bias band GEMM: P[l, c] = sum_d q[l,d] * rel[j0+c, d] ---
    const int j0 = r0 - l0 + (2048 - 16);   // (r - l) + 2047 at c = (r-r0)-(l-l0)+15
    v8f p[3];
#pragma unroll
    for (int t = 0; t < 3; ++t)
#pragma unroll
      for (int e = 0; e < 8; ++e) p[t][e] = 0.0f;
#pragma unroll
    for (int c = 0; c < 2; ++c)
#pragma unroll
      for (int t = 0; t < 3; ++t) {
        int j = j0 + t * 16 + lanen;
        j = j < 0 ? 0 : (j > RELROWS - 1 ? RELROWS - 1 : j);
        const bf16* rp = rel + (size_t)j * HD + c * 32;
        p[t] = wmma_bf16(aq[c], ld_frag(rp + kb0, rp + kb1), p[t]);
      }
#pragma unroll
    for (int t = 0; t < 3; ++t)
#pragma unroll
      for (int e = 0; e < 8; ++e)
        myPs[(e + 8 * half) * 48 + t * 16 + lanen] = p[t][e];

    // --- S = Q @ K^T (two 16-key halves) ---
    v8f s[2];
#pragma unroll
    for (int u = 0; u < 2; ++u)
#pragma unroll
      for (int e = 0; e < 8; ++e) s[u][e] = 0.0f;
#pragma unroll
    for (int c = 0; c < 2; ++c)
#pragma unroll
      for (int u = 0; u < 2; ++u) {
        const bf16* kk = Ks + (u * 16 + lanen) * 64 + c * 32;
        s[u] = wmma_bf16(aq[c], ld_frag(kk + kb0, kk + kb1), s[u]);
      }

    // scale + skewed rel gather: S += P[l, (r-r0)-(l-l0)+15]
#pragma unroll
    for (int u = 0; u < 2; ++u)
#pragma unroll
      for (int e = 0; e < 8; ++e) {
        const int mrl = e + 8 * half;
        const int c   = u * 16 + lanen - mrl + 15;   // in [0,46]
        s[u][e] = s[u][e] * 0.125f + myPs[mrl * 48 + c];
      }

    // --- online softmax (rows live on 16-lane groups; shuffle-reduce) ---
    float alpha[8];
#pragma unroll
    for (int e = 0; e < 8; ++e) {
      float t = fmaxf(s[0][e], s[1][e]);
#pragma unroll
      for (int msk = 8; msk >= 1; msk >>= 1) t = fmaxf(t, __shfl_xor(t, msk, 32));
      const float mn = fmaxf(mrow[e], t);
      alpha[e] = __expf(mrow[e] - mn);
      mrow[e]  = mn;
      const float p0 = __expf(s[0][e] - mn);
      const float p1 = __expf(s[1][e] - mn);
      s[0][e] = p0; s[1][e] = p1;
      float rs = p0 + p1;
#pragma unroll
      for (int msk = 8; msk >= 1; msk >>= 1) rs += __shfl_xor(rs, msk, 32);
      lsum[e] = lsum[e] * alpha[e] + rs;
    }

    // stage probabilities to LDS, reload in A-fragment layout as bf16
#pragma unroll
    for (int u = 0; u < 2; ++u)
#pragma unroll
      for (int e = 0; e < 8; ++e)
        myPs[(e + 8 * half) * 48 + u * 16 + lanen] = s[u][e];

    v16bf aatt;
    {
      const float* pr = myPs + lanen * 48;
#pragma unroll
      for (int i = 0; i < 8; ++i) {
        aatt[i]     = (bf16)pr[kb0 + i];
        aatt[i + 8] = (bf16)pr[kb1 + i];
      }
    }

    // rescale O and accumulate att @ V
#pragma unroll
    for (int nc = 0; nc < 4; ++nc)
#pragma unroll
      for (int e = 0; e < 8; ++e) oacc[nc][e] *= alpha[e];
#pragma unroll
    for (int nc = 0; nc < 4; ++nc) {
      const bf16* vb = Vt + (nc * 16 + lanen) * 32;
      oacc[nc] = wmma_bf16(aatt, ld_frag(vb + kb0, vb + kb1), oacc[nc]);
    }
  }

  // finalize: O /= l, write bf16 concat layout [B, L, H*64]
#pragma unroll
  for (int nc = 0; nc < 4; ++nc)
#pragma unroll
    for (int e = 0; e < 8; ++e) {
      const float val = oacc[nc][e] / lsum[e];
      const int   row = l0 + e + 8 * half;
      const int   d   = nc * 16 + lanen;
      o[((size_t)(b * LSEQ + row)) * DM + h * HD + d] = (bf16)val;
    }
}

// ---------------------------------------------------------------------------
// Launch
// ---------------------------------------------------------------------------
extern "C" void kernel_launch(void* const* d_in, const int* in_sizes, int n_in,
                              void* d_out, int out_size, void* d_ws, size_t ws_size,
                              hipStream_t stream) {
  const float* x   = (const float*)d_in[0];
  const float* Wq  = (const float*)d_in[1];
  const float* bq  = (const float*)d_in[2];
  const float* Wk  = (const float*)d_in[3];
  const float* bk  = (const float*)d_in[4];
  const float* Wv  = (const float*)d_in[5];
  const float* bv  = (const float*)d_in[6];
  const float* Wo  = (const float*)d_in[7];
  const float* bo  = (const float*)d_in[8];
  const float* rel = (const float*)d_in[9];

  const size_t nX = (size_t)GM * DM;      // 4,194,304
  const size_t nW = (size_t)DM * DM;      // 1,048,576
  const size_t nR = (size_t)RELROWS * HD; // 262,080

  char* ws = (char*)d_ws;
  bf16* xb   = (bf16*)ws; ws += nX * sizeof(bf16);
  bf16* wqb  = (bf16*)ws; ws += nW * sizeof(bf16);
  bf16* wkb  = (bf16*)ws; ws += nW * sizeof(bf16);
  bf16* wvb  = (bf16*)ws; ws += nW * sizeof(bf16);
  bf16* wob  = (bf16*)ws; ws += nW * sizeof(bf16);
  bf16* relb = (bf16*)ws; ws += ((nR + 63) & ~(size_t)63) * sizeof(bf16);
  bf16* qb   = (bf16*)ws; ws += nX * sizeof(bf16);
  bf16* kb_  = (bf16*)ws; ws += nX * sizeof(bf16);
  bf16* vb_  = (bf16*)ws; ws += nX * sizeof(bf16);
  bf16* ob   = (bf16*)ws; ws += nX * sizeof(bf16);

  auto cvt = [&](const float* s, bf16* d, size_t n) {
    cvt_f32_bf16<<<(unsigned)((n + 255) / 256), 256, 0, stream>>>(s, d, (int)n);
  };
  cvt(x,   xb,   nX);
  cvt(Wq,  wqb,  nW);
  cvt(Wk,  wkb,  nW);
  cvt(Wv,  wvb,  nW);
  cvt(Wo,  wob,  nW);
  cvt(rel, relb, nR);

  dim3 ggrid(DM / 128, GM / 128);   // (8, 32)
  gemm_wmma<0><<<ggrid, 256, 0, stream>>>(xb, wqb, bq, (void*)qb);
  gemm_wmma<0><<<ggrid, 256, 0, stream>>>(xb, wkb, bk, (void*)kb_);
  gemm_wmma<0><<<ggrid, 256, 0, stream>>>(xb, wvb, bv, (void*)vb_);

  dim3 agrid(LSEQ / 128, NH, BATCH); // (16, 16, 2)
  attn_wmma<<<agrid, 256, 0, stream>>>(qb, kb_, vb_, relb, ob);

  gemm_wmma<1><<<ggrid, 256, 0, stream>>>(ob, wob, bo, d_out);
}
